// EGATLayer_66907000537643
// MI455X (gfx1250) — matrix operand order repaired and checked by reading, not compiled, not run on hardware
//
#include <hip/hip_runtime.h>
#include <hip/hip_bf16.h>
#include <math.h>

typedef __attribute__((ext_vector_type(16))) __bf16 v16bf;
typedef __attribute__((ext_vector_type(8)))  float  v8f;

#define BB 4
#define NN 2048
#define NODE_IN 128
#define HH 64
#define EHD 8
#define ALPHA_LR 0.2f
#define NEG_INF_V (-9.0e15f)

union ABf16 { v16bf v; __bf16 s[16]; };
union Acc8  { v8f  v; float  s[8];  };

__device__ __forceinline__ float lrelu(float x) { return fmaxf(x, ALPHA_LR * x); } // branchless
__device__ __forceinline__ float elu1(float x)  { return x > 0.f ? x : __expf(x) - 1.f; }

// -------- Kernel 1: Wn = n @ W (fp32), store WnT in bf16 [b][h][i]; a1,a2 per node ----
__global__ void __launch_bounds__(HH)
k1_wn(const float* __restrict__ n, const float* __restrict__ W, const float* __restrict__ a,
      __bf16* __restrict__ wnT, float* __restrict__ a1, float* __restrict__ a2)
{
    const int node = blockIdx.x;            // 0 .. B*N-1
    const int h = threadIdx.x;              // 0..63
    const int b = node / NN, i = node % NN;
    const float* nrow = n + (size_t)node * NODE_IN;
    float acc = 0.f;
#pragma unroll 8
    for (int k = 0; k < NODE_IN; ++k) acc = fmaf(nrow[k], W[k * HH + h], acc);
    wnT[((size_t)b * HH + h) * NN + i] = (__bf16)acc;

    __shared__ float s1[HH], s2[HH];
    s1[h] = acc * a[h];
    s2[h] = acc * a[HH + h];
    __syncthreads();
    for (int off = HH / 2; off > 0; off >>= 1) {
        if (h < off) { s1[h] += s1[h + off]; s2[h] += s2[h + off]; }
        __syncthreads();
    }
    if (h == 0) { a1[node] = s1[0]; a2[node] = s2[0]; }
}

// -------- Kernel 2: per-row softmax stats: m, l = sum exp, s = sum exp*e --------------
__global__ void __launch_bounds__(256)
k2_stats(const float* __restrict__ e, const int* __restrict__ adj,
         const float* __restrict__ a1, const float* __restrict__ a2,
         const float* __restrict__ U, const float* __restrict__ a,
         float* __restrict__ rm, float* __restrict__ rl, float* __restrict__ rs)
{
    const int row = blockIdx.x;             // b*N + i
    const int t = threadIdx.x;
    __shared__ float lg[NN];
    __shared__ float ev[NN];
    __shared__ float red[256];

    float c = 0.f;
#pragma unroll
    for (int k = 0; k < EHD; ++k) c = fmaf(U[k], a[2 * HH + k], c);

    const float a1i = a1[row];
    const int b = row / NN;
    const float* erow = e + (size_t)row * NN;
    const int*   arow = adj + (size_t)row * NN;
    const float* a2b  = a2 + (size_t)b * NN;

    float mloc = -3.0e38f;
    for (int j = t; j < NN; j += 256) {
        __builtin_prefetch(erow + j + 256, 0, 3);      // global_prefetch_b8 (speculative-safe)
        float evj = erow[j];
        float x   = a1i + a2b[j] + c * evj;
        float l1  = lrelu(x);
        float msk = (arow[j] > 0) ? 1.0f : 0.0f;       // v_cndmask, no branch
        float lgt = fmaf(msk, l1, (1.0f - msk) * NEG_INF_V);  // exact blend
        ev[j] = evj; lg[j] = lgt;
        mloc = fmaxf(mloc, lgt);
    }
    red[t] = mloc; __syncthreads();
    for (int off = 128; off > 0; off >>= 1) {
        if (t < off) red[t] = fmaxf(red[t], red[t + off]);
        __syncthreads();
    }
    const float m = red[0];
    __syncthreads();

    float lsum = 0.f, ssum = 0.f;
    for (int j = t; j < NN; j += 256) {
        float p = __expf(lg[j] - m);
        lsum += p; ssum = fmaf(p, ev[j], ssum);
    }
    red[t] = lsum; __syncthreads();
    for (int off = 128; off > 0; off >>= 1) { if (t < off) red[t] += red[t + off]; __syncthreads(); }
    const float L = red[0];
    __syncthreads();
    red[t] = ssum; __syncthreads();
    for (int off = 128; off > 0; off >>= 1) { if (t < off) red[t] += red[t + off]; __syncthreads(); }
    if (t == 0) { rm[row] = m; rl[row] = L; rs[row] = red[0]; }
}

// -------- Kernel 3: node_part = softmax(att) @ Wn via v_wmma_f32_16x16x32_bf16 -------
// 4 independent waves per block; each wave owns one 16-row strip. p-tile is built in
// registers in contiguous K order (same as the B operand):
//   lanes 0-15 : row=lane,    K = {0..15} of the 32-wide step
//   lanes 16-31: row=lane-16, K = {16..31}
// B operand from WnT (bf16, [b][h][i]) : lane = column, contiguous 16 K values.
__global__ void __launch_bounds__(128)
k3_gemm(const float* __restrict__ e, const int* __restrict__ adj,
        const float* __restrict__ a1, const float* __restrict__ a2,
        const float* __restrict__ U, const float* __restrict__ a,
        const __bf16* __restrict__ wnT,
        const float* __restrict__ rm, const float* __restrict__ rl,
        float* __restrict__ out)
{
    const int lane = threadIdx.x & 31;
    const int wave = threadIdx.x >> 5;           // 0..3
    const int i0   = (blockIdx.x * 4 + wave) * 16;   // row strip
    const int b    = blockIdx.y;

    float c = 0.f;
#pragma unroll
    for (int k = 0; k < EHD; ++k) c = fmaf(U[k], a[2 * HH + k], c);

    const int row = lane & 15;
    const int hi  = lane >> 4;             // lane half
    const int gi  = b * NN + i0 + row;
    const float a1i  = a1[gi];
    const float mi   = rm[gi];
    const float argF = NEG_INF_V - mi;     // exp(argF): 0 normally, 1 for all-masked rows
    const float* erow = e   + (size_t)gi * NN;
    const int*   arow = adj + (size_t)gi * NN;
    const float* a2b  = a2  + (size_t)b * NN;
    const int col = lane & 15;
    const int ks  = hi * 16;               // B-operand per-lane K offset
    const int kb  = hi * 16;               // A-operand per-lane K offset
    const __bf16* wb = wnT + (size_t)b * HH * NN;

    Acc8 acc0, acc1, acc2, acc3;
    acc0.v = (v8f){}; acc1.v = (v8f){}; acc2.v = (v8f){}; acc3.v = (v8f){};

    for (int k0 = 0; k0 < NN; k0 += 32) {
        __builtin_prefetch(erow + k0 + 32, 0, 3);   // global_prefetch_b8
        __builtin_prefetch(arow + k0 + 32, 0, 3);

        // ---- per-lane slice of the p tile (16 values), fully branchless ----
        const float4* e4  = (const float4*)(erow + k0 + kb);
        const int4*   ad4 = (const int4*)  (arow + k0 + kb);
        const float4* s4  = (const float4*)(a2b  + k0 + kb);
        float4 ea = e4[0],  eb = e4[1],  ec = e4[2],  ed = e4[3];
        int4   ma = ad4[0], mb = ad4[1], mc = ad4[2], md = ad4[3];
        float4 sa = s4[0],  sb = s4[1],  sc = s4[2],  sd = s4[3];

        float ef[16] = {ea.x,ea.y,ea.z,ea.w, eb.x,eb.y,eb.z,eb.w,
                        ec.x,ec.y,ec.z,ec.w, ed.x,ed.y,ed.z,ed.w};
        int   af[16] = {ma.x,ma.y,ma.z,ma.w, mb.x,mb.y,mb.z,mb.w,
                        mc.x,mc.y,mc.z,mc.w, md.x,md.y,md.z,md.w};
        float sf[16] = {sa.x,sa.y,sa.z,sa.w, sb.x,sb.y,sb.z,sb.w,
                        sc.x,sc.y,sc.z,sc.w, sd.x,sd.y,sd.z,sd.w};

        ABf16 A;
#pragma unroll
        for (int t = 0; t < 16; ++t) {
            float x   = a1i + sf[t] + c * ef[t];
            float l1  = lrelu(x);                       // fmaxf: one VALU op
            float msk = (af[t] > 0) ? 1.0f : 0.0f;      // v_cndmask, no branch
            float arg = fmaf(msk, l1 - mi, (1.0f - msk) * argF);  // exact blend
            A.s[t] = (__bf16)__expf(arg);
        }

        // ---- B tiles: 4 column groups of 16 ----
        const size_t kidx = (size_t)(k0 + ks);
        v16bf B0 = *(const v16bf*)(wb + ((size_t)(col      )) * NN + kidx);
        v16bf B1 = *(const v16bf*)(wb + ((size_t)(col + 16 )) * NN + kidx);
        v16bf B2 = *(const v16bf*)(wb + ((size_t)(col + 32 )) * NN + kidx);
        v16bf B3 = *(const v16bf*)(wb + ((size_t)(col + 48 )) * NN + kidx);

        acc0.v = __builtin_amdgcn_wmma_f32_16x16x32_bf16(false, A.v, false, B0, (short)0, acc0.v, false, false);
        acc1.v = __builtin_amdgcn_wmma_f32_16x16x32_bf16(false, A.v, false, B1, (short)0, acc1.v, false, false);
        acc2.v = __builtin_amdgcn_wmma_f32_16x16x32_bf16(false, A.v, false, B2, (short)0, acc2.v, false, false);
        acc3.v = __builtin_amdgcn_wmma_f32_16x16x32_bf16(false, A.v, false, B3, (short)0, acc3.v, false, false);
    }

    // ---- scale by 1/l, ELU, store (C layout: VGPR v -> M = v + hi*8, N = col) ----
    const float* rlb = rl + (size_t)b * NN;
#pragma unroll
    for (int v = 0; v < 8; ++v) {
        const int r  = v + hi * 8;
        const int oi = b * NN + i0 + r;
        const float inv = 1.0f / rlb[i0 + r];
        float* orow = out + (size_t)oi * (HH + EHD);
        orow[ 0 + col] = elu1(acc0.s[v] * inv);
        orow[16 + col] = elu1(acc1.s[v] * inv);
        orow[32 + col] = elu1(acc2.s[v] * inv);
        orow[48 + col] = elu1(acc3.s[v] * inv);
    }
}

// -------- Kernel 4: edge part: elu((s/l) * U[k]) into out cols 64..71 -----------------
__global__ void __launch_bounds__(256)
k4_edge(const float* __restrict__ U, const float* __restrict__ rl,
        const float* __restrict__ rs, float* __restrict__ out)
{
    int idx = blockIdx.x * blockDim.x + threadIdx.x;   // B*N*EH
    if (idx >= BB * NN * EHD) return;
    int k = idx & (EHD - 1), row = idx / EHD;
    float v = (rs[row] / rl[row]) * U[k];
    out[(size_t)row * (HH + EHD) + HH + k] = elu1(v);
}

extern "C" void kernel_launch(void* const* d_in, const int* in_sizes, int n_in,
                              void* d_out, int out_size, void* d_ws, size_t ws_size,
                              hipStream_t stream)
{
    (void)in_sizes; (void)n_in; (void)out_size; (void)ws_size;
    const float* n   = (const float*)d_in[0];
    const float* e   = (const float*)d_in[1];
    const int*   adj = (const int*)d_in[2];
    const float* W   = (const float*)d_in[3];
    const float* U   = (const float*)d_in[4];
    const float* a   = (const float*)d_in[5];
    float* out = (float*)d_out;

    char* ws = (char*)d_ws;
    __bf16* wnT = (__bf16*)ws;        ws += (size_t)BB * HH * NN * sizeof(__bf16);
    float* a1 = (float*)ws;           ws += (size_t)BB * NN * sizeof(float);
    float* a2 = (float*)ws;           ws += (size_t)BB * NN * sizeof(float);
    float* rm = (float*)ws;           ws += (size_t)BB * NN * sizeof(float);
    float* rl = (float*)ws;           ws += (size_t)BB * NN * sizeof(float);
    float* rs = (float*)ws;           ws += (size_t)BB * NN * sizeof(float);

    k1_wn   <<<dim3(BB * NN), dim3(HH),  0, stream>>>(n, W, a, wnT, a1, a2);
    k2_stats<<<dim3(BB * NN), dim3(256), 0, stream>>>(e, adj, a1, a2, U, a, rm, rl, rs);
    k3_gemm <<<dim3(NN / 64, BB), dim3(128), 0, stream>>>(e, adj, a1, a2, U, a, wnT, rm, rl, out);
    k4_edge <<<dim3((BB * NN * EHD + 255) / 256), dim3(256), 0, stream>>>(U, rl, rs, out);
}
